// GNN_clf_8916352106605
// MI455X (gfx1250) — matrix operand order repaired and checked by reading, not compiled, not run on hardware
//
#include <hip/hip_runtime.h>
#include <hip/hip_bf16.h>
#include <math.h>

typedef __attribute__((ext_vector_type(16))) _Float16 v16h;
typedef __attribute__((ext_vector_type(8)))  _Float16 v8h;
typedef __attribute__((ext_vector_type(8)))  float    v8f;

#define M_ROWS 32768   // B * N = 16 * 2048
#define NBATCH 16
#define NNODE  2048
#define FD     128

// ---------------------------------------------------------------------------
// Weight packing: f32 row-major W[K][N] -> f16 WMMA B fragments.
// Fragment (kb, nb) covers W[kb*32 .. kb*32+31][nb*16 .. nb*16+15].
// Per ISA 7.12.2 (B 32x16, 16-bit): lane L holds column n = nb*16 + (L&15);
// lanes 0-15 hold K = kb*32 + {0..7, 16..23}, lanes 16-31 hold
// K = kb*32 + {8..15, 24..31}. Stored so the GEMM reads 32 contiguous bytes per lane:
//   Wp[(nb*(K/32)+kb)*512 + L*16 + hh]
// ---------------------------------------------------------------------------
__global__ void pack_weight_kernel(const float* __restrict__ W,
                                   _Float16* __restrict__ Wp,
                                   int K, int N) {
  int idx = blockIdx.x * 256 + threadIdx.x;
  if (idx >= K * N) return;
  int fragsK = K >> 5;
  int nb   = idx / (fragsK * 512);
  int rem  = idx % (fragsK * 512);
  int kb   = rem >> 9;
  int r2   = rem & 511;
  int lane = r2 >> 4;
  int hh   = r2 & 15;
  int n = nb * 16 + (lane & 15);
  int k = kb * 32 + ((lane >= 16) ? 8 : 0) + (hh & 7) + ((hh >> 3) << 4);
  Wp[idx] = (_Float16)W[k * N + n];
}

__global__ void f32_to_f16_kernel(const float* __restrict__ in,
                                  _Float16* __restrict__ out, int n) {
  int i = blockIdx.x * 256 + threadIdx.x;
  if (i < n) out[i] = (_Float16)in[i];
}

// ---------------------------------------------------------------------------
// WMMA GEMM: Out[M x NOUT] = relu(A[M x K] @ W + bias), f16 in/out, f32 accum.
// One wave computes a 32(M) x 64(N) tile: 2 row tiles x 4 col tiles = 8
// accumulators. Each B fragment is reused by 2 WMMAs (halves weight loads,
// doubles independent WMMA work between load waits).
// A per-lane layout (ISA 7.12.2, 16-bit A 16x32): lane&15 = row; lanes 0-15
// hold K halves {0..7, 16..23}, lanes 16-31 hold {8..15, 24..31}.
// ---------------------------------------------------------------------------
template<int K, int NOUT>
__global__ __launch_bounds__(128) void gemm_bias_relu(
    const _Float16* __restrict__ A,
    const _Float16* __restrict__ Wp,
    const float*    __restrict__ bias,
    _Float16*       __restrict__ Out) {
  constexpr int TILES_N = NOUT / 64;
  const int wid  = threadIdx.x >> 5;
  const int lane = threadIdx.x & 31;
  const int gw   = blockIdx.x * 4 + wid;
  const int rowPair = gw / TILES_N;       // 32-row super-tile
  const int nT      = gw % TILES_N;
  const int M0   = rowPair * 32;
  const int hiK  = (lane >> 4) * 8;       // upper half-wave reads K+8 chunks

  v8f acc[2][4] = {{v8f{0.f}, v8f{0.f}, v8f{0.f}, v8f{0.f}},
                   {v8f{0.f}, v8f{0.f}, v8f{0.f}, v8f{0.f}}};
  const _Float16* ap0 = A + (size_t)(M0 + (lane & 15)) * K;
  const _Float16* ap1 = ap0 + (size_t)16 * K;

  #pragma unroll
  for (int kb = 0; kb < K / 32; ++kb) {
    union { v16h v; v8h h[2]; } a0, a1;
    a0.h[0] = *(const v8h*)(ap0 + kb * 32 + hiK);
    a0.h[1] = *(const v8h*)(ap0 + kb * 32 + 16 + hiK);
    a1.h[0] = *(const v8h*)(ap1 + kb * 32 + hiK);
    a1.h[1] = *(const v8h*)(ap1 + kb * 32 + 16 + hiK);
    #pragma unroll
    for (int nt = 0; nt < 4; ++nt) {
      const int nb = nT * 4 + nt;
      const _Float16* bp = Wp + ((size_t)(nb * (K / 32) + kb)) * 512 + lane * 16;
      union { v16h v; v8h h[2]; } b;
      b.h[0] = *(const v8h*)(bp);
      b.h[1] = *(const v8h*)(bp + 8);
      acc[0][nt] = __builtin_amdgcn_wmma_f32_16x16x32_f16(
          false, a0.v, false, b.v, (short)0, acc[0][nt], false, false);
      acc[1][nt] = __builtin_amdgcn_wmma_f32_16x16x32_f16(
          false, a1.v, false, b.v, (short)0, acc[1][nt], false, false);
    }
  }

  // C/D layout: lane&15 = column; lanes 0-15 -> rows +0..7 (vgpr r),
  // lanes 16-31 -> rows +8..15.
  #pragma unroll
  for (int rt = 0; rt < 2; ++rt) {
    const int mBase = M0 + rt * 16 + (lane >> 4) * 8;
    #pragma unroll
    for (int nt = 0; nt < 4; ++nt) {
      const int n = (nT * 4 + nt) * 16 + (lane & 15);
      const float bv = bias[n];
      #pragma unroll
      for (int r = 0; r < 8; ++r) {
        float v = acc[rt][nt][r] + bv;
        v = v > 0.0f ? v : 0.0f;
        Out[(size_t)(mBase + r) * NOUT + n] = (_Float16)v;
      }
    }
  }
}

// ---------------------------------------------------------------------------
// Per-batch reduction of edge activations: [16][2048][128] -> [16][16][128]
// partials (128 rows each), then fold 16 partials -> [16][128].
// ---------------------------------------------------------------------------
__global__ void edge_partial_kernel(const _Float16* __restrict__ edge,
                                    float* __restrict__ partial) {
  int b = blockIdx.x, c = blockIdx.y, f = threadIdx.x;
  const _Float16* p = edge + ((size_t)b * NNODE + (size_t)c * 128) * FD + f;
  float s = 0.0f;
  for (int i = 0; i < 128; ++i) s += (float)p[(size_t)i * FD];
  partial[(b * 16 + c) * FD + f] = s;
}

__global__ void esum_final_kernel(const float* __restrict__ partial,
                                  float* __restrict__ esum) {
  int b = blockIdx.x, f = threadIdx.x;
  float s = 0.0f;
  for (int c = 0; c < 16; ++c) s += partial[(b * 16 + c) * FD + f];
  esum[b * FD + f] = s;
}

// h[b][i][f] = (node + esum[b][f] - edge) / N
__global__ void combine_kernel(const _Float16* __restrict__ node,
                               const _Float16* __restrict__ edge,
                               const float* __restrict__ esum,
                               _Float16* __restrict__ h) {
  size_t i = (size_t)blockIdx.x * 256 + threadIdx.x;
  if (i >= (size_t)M_ROWS * FD) return;
  int f = (int)(i & (FD - 1));
  int b = (int)(i >> 18);              // / (2048*128)
  float v = ((float)node[i] + esum[b * FD + f] - (float)edge[i]) *
            (1.0f / (float)NNODE);
  h[i] = (_Float16)v;
}

// Final 128 -> 1 layer: sigmoid(dot(act_row, W) + b)
__global__ void out_final_kernel(const _Float16* __restrict__ act,
                                 const float* __restrict__ W,
                                 const float* __restrict__ bptr,
                                 float* __restrict__ out) {
  __shared__ float w[FD];
  if (threadIdx.x < FD) w[threadIdx.x] = W[threadIdx.x];
  __syncthreads();
  int row = blockIdx.x * 256 + threadIdx.x;
  if (row >= M_ROWS) return;
  const _Float16* a = act + (size_t)row * FD;
  float s = bptr[0];
  #pragma unroll 8
  for (int i = 0; i < FD; ++i) s += (float)a[i] * w[i];
  out[row] = 1.0f / (1.0f + __expf(-s));
}

// ---------------------------------------------------------------------------
extern "C" void kernel_launch(void* const* d_in, const int* in_sizes, int n_in,
                              void* d_out, int out_size, void* d_ws, size_t ws_size,
                              hipStream_t stream) {
  (void)in_sizes; (void)n_in; (void)out_size; (void)ws_size;
  // dict order: X, A, node_params (W0,b0..W3,b3), edge_params, out_params
  const float* X = (const float*)d_in[0];
  const float* nW[4]  = {(const float*)d_in[2],  (const float*)d_in[4],
                         (const float*)d_in[6],  (const float*)d_in[8]};
  const float* nBv[4] = {(const float*)d_in[3],  (const float*)d_in[5],
                         (const float*)d_in[7],  (const float*)d_in[9]};
  const float* eW[4]  = {(const float*)d_in[10], (const float*)d_in[12],
                         (const float*)d_in[14], (const float*)d_in[16]};
  const float* eBv[4] = {(const float*)d_in[11], (const float*)d_in[13],
                         (const float*)d_in[15], (const float*)d_in[17]};
  const float* oW[4]  = {(const float*)d_in[18], (const float*)d_in[20],
                         (const float*)d_in[22], (const float*)d_in[24]};
  const float* oBv[4] = {(const float*)d_in[19], (const float*)d_in[21],
                         (const float*)d_in[23], (const float*)d_in[25]};

  char* ws = (char*)d_ws;
  auto alloc = [&](size_t bytes) -> char* {
    char* p = ws;
    ws += (bytes + 255) & ~(size_t)255;
    return p;
  };
  _Float16* h16   = (_Float16*)alloc((size_t)M_ROWS * 128 * 2);
  _Float16* a128  = (_Float16*)alloc((size_t)M_ROWS * 128 * 2);
  _Float16* a256  = (_Float16*)alloc((size_t)M_ROWS * 256 * 2);
  _Float16* b128  = (_Float16*)alloc((size_t)M_ROWS * 128 * 2);
  _Float16* nodeO = (_Float16*)alloc((size_t)M_ROWS * 128 * 2);
  _Float16* edgeO = (_Float16*)alloc((size_t)M_ROWS * 128 * 2);
  float* partial  = (float*)alloc((size_t)16 * 16 * FD * 4);
  float* esum     = (float*)alloc((size_t)16 * FD * 4);

  const int dims[4][2] = {{128, 128}, {128, 256}, {256, 128}, {128, 128}};
  _Float16 *pN[4], *pE[4], *pO[3];
  for (int l = 0; l < 4; ++l) pN[l] = (_Float16*)alloc((size_t)dims[l][0] * dims[l][1] * 2);
  for (int l = 0; l < 4; ++l) pE[l] = (_Float16*)alloc((size_t)dims[l][0] * dims[l][1] * 2);
  for (int l = 0; l < 3; ++l) pO[l] = (_Float16*)alloc((size_t)dims[l][0] * dims[l][1] * 2);

  // 1) pack weights to f16 WMMA fragment layout
  for (int l = 0; l < 4; ++l) {
    int t = dims[l][0] * dims[l][1];
    pack_weight_kernel<<<(t + 255) / 256, 256, 0, stream>>>(nW[l], pN[l], dims[l][0], dims[l][1]);
    pack_weight_kernel<<<(t + 255) / 256, 256, 0, stream>>>(eW[l], pE[l], dims[l][0], dims[l][1]);
  }
  for (int l = 0; l < 3; ++l) {
    int t = dims[l][0] * dims[l][1];
    pack_weight_kernel<<<(t + 255) / 256, 256, 0, stream>>>(oW[l], pO[l], dims[l][0], dims[l][1]);
  }

  // 2) X -> f16
  {
    int t = M_ROWS * FD;
    f32_to_f16_kernel<<<(t + 255) / 256, 256, 0, stream>>>(X, h16, t);
  }

  // grid: (M/32 row-supertiles) * (NOUT/64 col-tiles) waves, 4 waves/block
  // <128,128>: 1024*2/4 = 512 blocks; <128,256>: 1024*4/4 = 1024; <256,128>: 512
  auto run_mlp4 = [&](_Float16* in, _Float16* const p[4],
                      const float* const bv[4], _Float16* out) {
    gemm_bias_relu<128, 128><<< 512, 128, 0, stream>>>(in,   p[0], bv[0], a128);
    gemm_bias_relu<128, 256><<<1024, 128, 0, stream>>>(a128, p[1], bv[1], a256);
    gemm_bias_relu<256, 128><<< 512, 128, 0, stream>>>(a256, p[2], bv[2], b128);
    gemm_bias_relu<128, 128><<< 512, 128, 0, stream>>>(b128, p[3], bv[3], out);
  };

  // 3) three message-passing steps
  for (int step = 0; step < 3; ++step) {
    run_mlp4(h16, pN, nBv, nodeO);
    run_mlp4(h16, pE, eBv, edgeO);
    edge_partial_kernel<<<dim3(16, 16), 128, 0, stream>>>(edgeO, partial);
    esum_final_kernel<<<16, 128, 0, stream>>>(partial, esum);
    combine_kernel<<<(M_ROWS * FD) / 256, 256, 0, stream>>>(nodeO, edgeO, esum, h16);
  }

  // 4) output MLP (relu on first 3 layers), then dot + sigmoid
  gemm_bias_relu<128, 128><<< 512, 128, 0, stream>>>(h16,  pO[0], oBv[0], a128);
  gemm_bias_relu<128, 256><<<1024, 128, 0, stream>>>(a128, pO[1], oBv[1], a256);
  gemm_bias_relu<256, 128><<< 512, 128, 0, stream>>>(a256, pO[2], oBv[2], b128);
  out_final_kernel<<<M_ROWS / 256, 256, 0, stream>>>(b128, oW[3], oBv[3], (float*)d_out);
}
